// NLRWDense_52355651338916
// MI455X (gfx1250) — matrix operand order repaired and checked by reading, not compiled, not run on hardware
//
#include <hip/hip_runtime.h>
#include <hip/hip_bf16.h>

#define DPC   0.05f
#define NR    4096      // n rows of x
#define DD    1024      // feature dim
#define MK    512       // kernel rows
#define NITER 96        // Neumann iterations: contraction ~0.8 => 0.8^96 ~ 5e-10

typedef __attribute__((ext_vector_type(16))) __bf16 v16bf;
typedef __attribute__((ext_vector_type(8)))  __bf16 v8bf;
typedef __attribute__((ext_vector_type(8)))  float  v8f;

// Register tile per wave: (16*MT) rows x (16*NT) cols
#define MT 4
#define NT 2
#define WTM (16 * MT)   // 64
#define WTN (16 * NT)   // 32
#define BTM (2 * WTM)   // 128  (2x2 waves per block)
#define BTN (2 * WTN)   // 64

// ---------------------------------------------------------------- WMMA helpers

static __device__ inline v8f wmma_bf16(v16bf a, v16bf b, v8f c) {
  // D = A(16x32 bf16) * B(32x16 bf16) + C(16x16 f32)
  return __builtin_amdgcn_wmma_f32_16x16x32_bf16(
      /*neg_a=*/false, a, /*neg_b=*/false, b,
      /*c_mod=*/(short)0, c, /*reuse_a=*/false, /*reuse_b=*/false);
}

// A operand: row-major [rows x lda] bf16. ISA 16-bit A layout:
// lanes 0-15: M=lane, elems = K {0..7, 16..23}; lanes 16-31: M=lane-16, K {8..15, 24..31}
static __device__ inline v16bf load_a_frag(const __bf16* A, int lda, int row0,
                                           int k0, int lane) {
  int m    = lane & 15;
  int koff = (lane & 16) ? 8 : 0;
  const __bf16* p = A + (size_t)(row0 + m) * lda + (k0 + koff);
  v8bf lo = *(const v8bf*)(p);        // K = k0+koff .. +7
  v8bf hi = *(const v8bf*)(p + 16);   // K = k0+koff+16 .. +23
  v16bf f;
#pragma unroll
  for (int i = 0; i < 8; ++i) { f[i] = lo[i]; f[i + 8] = hi[i]; }
  return f;
}

// B operand from row-major source S where B[k][n] = S[col0+n][k].
// B layout (matches A): lanes 0-15: N=lane, K {0..7, 16..23}; lanes 16-31: N=lane-16, K {8..15, 24..31}
// -> two b128 loads per lane.
static __device__ inline v16bf load_b_frag(const __bf16* S, int lds_, int col0,
                                           int k0, int lane) {
  int n    = lane & 15;
  int koff = (lane & 16) ? 8 : 0;
  const __bf16* p = S + (size_t)(col0 + n) * lds_ + (k0 + koff);
  v8bf lo = *(const v8bf*)(p);
  v8bf hi = *(const v8bf*)(p + 16);
  v16bf f;
#pragma unroll
  for (int i = 0; i < 8; ++i) { f[i] = lo[i]; f[i + 8] = hi[i]; }
  return f;
}

// Unified register-tiled GEMM core: C[m][n] += A[i0+16m.., :K] * S[n0+16n.., :K]^T
// A row-major [*, lda]; S row-major [*, ldb] supplying B columns.
static __device__ inline void gemm_core(const __bf16* __restrict__ A, int lda,
                                        const __bf16* __restrict__ S, int ldb,
                                        int i0, int n0, int K, int lane,
                                        bool pf, v8f c[MT][NT]) {
  for (int k = 0; k < K; k += 32) {
    if (pf && (k + 32 < K)) {  // gfx1250 global_prefetch_b8 for next K tile
      __builtin_prefetch(A + (size_t)(i0 + (lane & 15)) * lda + k + 32, 0, 1);
      __builtin_prefetch(S + (size_t)(n0 + (lane & 15)) * ldb + k + 32, 0, 1);
    }
    v16bf a[MT];
#pragma unroll
    for (int m = 0; m < MT; ++m) a[m] = load_a_frag(A, lda, i0 + 16 * m, k, lane);
#pragma unroll
    for (int n = 0; n < NT; ++n) {
      v16bf b = load_b_frag(S, ldb, n0 + 16 * n, k, lane);
#pragma unroll
      for (int m = 0; m < MT; ++m) c[m][n] = wmma_bf16(a[m], b, c[m][n]);
    }
  }
}

// ---------------------------------------------------------------- prep kernels

// bf16 convert + exact f32 row-norms. One block per row.
__global__ __launch_bounds__(256) void cvt_norm_kernel(const float* __restrict__ src,
                                                       __bf16* __restrict__ dst,
                                                       float* __restrict__ norm,
                                                       int cols) {
  int row = blockIdx.x, tid = threadIdx.x;
  const float* s = src + (size_t)row * cols;
  __bf16*      d = dst + (size_t)row * cols;
  float acc = 0.0f;
  for (int c = tid; c < cols; c += 256) {
    float v = s[c];
    d[c] = (__bf16)v;
    acc += v * v;
  }
  __shared__ float red[256];
  red[tid] = acc;
  __syncthreads();
  for (int off = 128; off > 0; off >>= 1) {
    if (tid < off) red[tid] += red[tid + off];
    __syncthreads();
  }
  if (tid == 0) norm[row] = red[0];
}

// ---------------------------------------------------------------- affinity GEMMs

// Tuu[i][j] = (i==j) ? 0 : exp(-dp*sqrt(max(2*nb[i] - 2*x_i.x_j, 0)))   (bf16 out)
__global__ __launch_bounds__(128) void tuu_kernel(const __bf16* __restrict__ xb,
                                                  const float* __restrict__ nb,
                                                  __bf16* __restrict__ Tuu) {
  int lane = threadIdx.x, wy = threadIdx.y;
  int i0 = blockIdx.y * BTM + (wy >> 1) * WTM;
  int j0 = blockIdx.x * BTN + (wy & 1) * WTN;
  v8f c[MT][NT] = {};
  gemm_core(xb, DD, xb, DD, i0, j0, DD, lane, false, c);

  int ln = lane & 15, hi8 = (lane & 16) ? 8 : 0;
#pragma unroll
  for (int m = 0; m < MT; ++m)
#pragma unroll
    for (int n = 0; n < NT; ++n)
#pragma unroll
      for (int r = 0; r < 8; ++r) {
        int gi = i0 + 16 * m + r + hi8, gj = j0 + 16 * n + ln;
        float Dv = fmaxf(2.0f * nb[gi] - 2.0f * c[m][n][r], 0.0f);
        float tv = (gi == gj) ? 0.0f : __expf(-DPC * sqrtf(Dv));
        Tuu[(size_t)gi * NR + gj] = (__bf16)tv;
      }
}

// Tul[i][l] = exp(-dp*sqrt(max(nb[i] + na[l] - 2*x_i.k_l, 0)))   (f32 out)
__global__ __launch_bounds__(128) void tul_kernel(const __bf16* __restrict__ xb,
                                                  const __bf16* __restrict__ kb,
                                                  const float* __restrict__ nb,
                                                  const float* __restrict__ na,
                                                  float* __restrict__ Tul) {
  int lane = threadIdx.x, wy = threadIdx.y;
  int i0 = blockIdx.y * BTM + (wy >> 1) * WTM;
  int j0 = blockIdx.x * BTN + (wy & 1) * WTN;
  v8f c[MT][NT] = {};
  gemm_core(xb, DD, kb, DD, i0, j0, DD, lane, false, c);

  int ln = lane & 15, hi8 = (lane & 16) ? 8 : 0;
#pragma unroll
  for (int m = 0; m < MT; ++m)
#pragma unroll
    for (int n = 0; n < NT; ++n)
#pragma unroll
      for (int r = 0; r < 8; ++r) {
        int gi = i0 + 16 * m + r + hi8, gj = j0 + 16 * n + ln;
        float Dv = fmaxf(nb[gi] + na[gj] - 2.0f * c[m][n][r], 0.0f);
        Tul[(size_t)gi * MK + gj] = __expf(-DPC * sqrtf(Dv));
      }
}

// ---------------------------------------------------------------- normalization

__global__ __launch_bounds__(256) void rowsum_kernel(const __bf16* __restrict__ Tuu,
                                                     const float* __restrict__ Tul,
                                                     float* __restrict__ invSum) {
  int row = blockIdx.x, tid = threadIdx.x;
  float s = 0.0f;
  const __bf16* pu = Tuu + (size_t)row * NR;
  for (int j = tid; j < NR; j += 256) s += (float)pu[j];
  const float* pl = Tul + (size_t)row * MK;
  for (int j = tid; j < MK; j += 256) s += pl[j];
  __shared__ float red[256];
  red[tid] = s;
  __syncthreads();
  for (int off = 128; off > 0; off >>= 1) {
    if (tid < off) red[tid] += red[tid + off];
    __syncthreads();
  }
  if (tid == 0) invSum[row] = 1.0f / red[0];
}

// Puu = Tuu * invSum[row]  (in place, bf16)
__global__ __launch_bounds__(256) void scale_uu_kernel(__bf16* __restrict__ Tuu,
                                                       const float* __restrict__ invSum) {
  size_t stride = (size_t)gridDim.x * blockDim.x;
  for (size_t idx = (size_t)blockIdx.x * blockDim.x + threadIdx.x;
       idx < (size_t)NR * NR; idx += stride) {
    int row = (int)(idx >> 12);   // / 4096
    Tuu[idx] = (__bf16)((float)Tuu[idx] * invSum[row]);
  }
}

// Pul = Tul * invSum (in place, f32) and Y0^T = bf16(Pul) (transposed for B-operand loads)
__global__ __launch_bounds__(256) void scale_ul_kernel(float* __restrict__ Tul,
                                                       const float* __restrict__ invSum,
                                                       __bf16* __restrict__ Yt0) {
  size_t stride = (size_t)gridDim.x * blockDim.x;
  for (size_t idx = (size_t)blockIdx.x * blockDim.x + threadIdx.x;
       idx < (size_t)NR * MK; idx += stride) {
    int i = (int)(idx >> 9);      // / 512
    int l = (int)(idx & (MK - 1));
    float p = Tul[idx] * invSum[i];
    Tul[idx] = p;
    Yt0[(size_t)l * NR + i] = (__bf16)p;
  }
}

// ---------------------------------------------------------------- Neumann iteration

// Ynew[i][n] = sum_k Puu[i][k]*Yt[n][k] + Pul[i][n]; write bf16 Yt_next and f32 Yf.
__global__ __launch_bounds__(128) void neumann_kernel(const __bf16* __restrict__ Puu,
                                                      const __bf16* __restrict__ Ytc,
                                                      const float* __restrict__ Pul,
                                                      __bf16* __restrict__ Ytn,
                                                      float* __restrict__ Yf) {
  int lane = threadIdx.x, wy = threadIdx.y;
  int i0 = blockIdx.y * BTM + (wy >> 1) * WTM;
  int n0 = blockIdx.x * BTN + (wy & 1) * WTN;
  v8f c[MT][NT] = {};
  gemm_core(Puu, NR, Ytc, NR, i0, n0, NR, lane, true, c);

  int ln = lane & 15, hi8 = (lane & 16) ? 8 : 0;
#pragma unroll
  for (int m = 0; m < MT; ++m)
#pragma unroll
    for (int n = 0; n < NT; ++n)
#pragma unroll
      for (int r = 0; r < 8; ++r) {
        int gi = i0 + 16 * m + r + hi8, gn = n0 + 16 * n + ln;
        float v = c[m][n][r] + Pul[(size_t)gi * MK + gn];
        Ytn[(size_t)gn * NR + gi] = (__bf16)v;   // transposed bf16 for next iter B loads
        Yf[(size_t)gi * MK + gn]  = v;           // row-major f32 result
      }
}

__global__ __launch_bounds__(256) void finalize_kernel(const float* __restrict__ Yf,
                                                       float* __restrict__ out) {
  size_t stride = (size_t)gridDim.x * blockDim.x;
  for (size_t idx = (size_t)blockIdx.x * blockDim.x + threadIdx.x;
       idx < (size_t)NR * MK; idx += stride)
    out[idx] = fmaxf(Yf[idx], 0.0f);
}

// ---------------------------------------------------------------- launch

extern "C" void kernel_launch(void* const* d_in, const int* in_sizes, int n_in,
                              void* d_out, int out_size, void* d_ws, size_t ws_size,
                              hipStream_t stream) {
  const float* x   = (const float*)d_in[0];   // [4096,1024]
  const float* krn = (const float*)d_in[1];   // [512,1024]
  float* out = (float*)d_out;                 // [4096,512]

  char* ws = (char*)d_ws;
  size_t off = 0;
  auto alloc = [&](size_t bytes) -> void* {
    void* p = ws + off;
    off = (off + bytes + 255) & ~(size_t)255;
    return p;
  };
  __bf16* xb     = (__bf16*)alloc((size_t)NR * DD * 2);   // 8 MiB
  __bf16* kb     = (__bf16*)alloc((size_t)MK * DD * 2);   // 1 MiB
  float*  nb     = (float*) alloc((size_t)NR * 4);
  float*  na     = (float*) alloc((size_t)MK * 4);
  __bf16* Tuu    = (__bf16*)alloc((size_t)NR * NR * 2);   // 32 MiB (becomes Puu, L2-resident)
  float*  Tul    = (float*) alloc((size_t)NR * MK * 4);   // 8 MiB (becomes Pul)
  float*  invSum = (float*) alloc((size_t)NR * 4);
  __bf16* Yt0    = (__bf16*)alloc((size_t)MK * NR * 2);   // 4 MiB
  __bf16* Yt1    = (__bf16*)alloc((size_t)MK * NR * 2);   // 4 MiB
  float*  Yf     = (float*) alloc((size_t)NR * MK * 4);   // 8 MiB

  dim3 wblk(32, 4);  // 4 wave32s per block, 2x2 arrangement of 64x32 wave tiles

  cvt_norm_kernel<<<NR, 256, 0, stream>>>(x, xb, nb, DD);
  cvt_norm_kernel<<<MK, 256, 0, stream>>>(krn, kb, na, DD);

  tuu_kernel<<<dim3(NR / BTN, NR / BTM), wblk, 0, stream>>>(xb, nb, Tuu);
  tul_kernel<<<dim3(MK / BTN, NR / BTM), wblk, 0, stream>>>(xb, kb, nb, na, Tul);

  rowsum_kernel<<<NR, 256, 0, stream>>>(Tuu, Tul, invSum);
  scale_uu_kernel<<<4096, 256, 0, stream>>>(Tuu, invSum);
  scale_ul_kernel<<<2048, 256, 0, stream>>>(Tul, invSum, Yt0);

  __bf16* cur = Yt0;
  __bf16* nxt = Yt1;
  for (int it = 0; it < NITER; ++it) {
    neumann_kernel<<<dim3(MK / BTN, NR / BTM), wblk, 0, stream>>>(Tuu, cur, Tul, nxt, Yf);
    __bf16* t = cur; cur = nxt; nxt = t;
  }
  finalize_kernel<<<2048, 256, 0, stream>>>(Yf, out);
}
